// TransformerEncoderLayer_72808285602449
// MI455X (gfx1250) — compile-verified, matrix operand order checked by reading, not run on hardware
//
#include <hip/hip_runtime.h>
#include <hip/hip_bf16.h>

// ---------------------------------------------------------------------------
// Transformer encoder layer for MI455X (gfx1250), wave32 + WMMA f16.
// B=8, S=1024 (32x32), C=256, N_HEAD=8, D_K=256 (per head!), D_FF=2048.
// GEMMs: double-buffered LDS tiles staged with GLOBAL_LOAD_ASYNC_TO_LDS_B128
// (ASYNCcnt) when the toolchain exposes the builtins.
// ---------------------------------------------------------------------------

typedef _Float16 half8 __attribute__((ext_vector_type(8)));
typedef _Float16 v16h  __attribute__((ext_vector_type(16)));
typedef float    v8f   __attribute__((ext_vector_type(8)));

#define NB   8
#define NS   1024
#define NC   256
#define NH   8
#define DK   256
#define HDK  2048
#define DFF  2048
#define NM   (NB * NS)   // 8192 rows

// ---------------------------------------------------------------------------
// Async global->LDS copy (16B per lane), CDNA5 ASYNCcnt path with fallback.
// Builtin signature (from clang diagnostic): first param is
//   int __attribute__((vector_size(16)))  addrspace(1) *
// ---------------------------------------------------------------------------
#if defined(__AMDGCN__) && \
    __has_builtin(__builtin_amdgcn_global_load_async_to_lds_b128) && \
    __has_builtin(__builtin_amdgcn_s_wait_asynccnt)
#define USE_ASYNC_LDS 1
#else
#define USE_ASYNC_LDS 0
#endif

#if USE_ASYNC_LDS
typedef int vint4 __attribute__((vector_size(16)));
typedef __attribute__((address_space(1))) vint4* as1_vint4_ptr;
typedef __attribute__((address_space(3))) vint4* as3_vint4_ptr;

__device__ __forceinline__ void g2l_b128(const _Float16* g, _Float16* l) {
    __builtin_amdgcn_global_load_async_to_lds_b128(
        (as1_vint4_ptr)(void*)const_cast<_Float16*>(g),
        (as3_vint4_ptr)(void*)l,
        /*offset=*/0, /*cpol=*/0);
}
__device__ __forceinline__ void wait_async() {
    __builtin_amdgcn_s_wait_asynccnt(0);
}
#else
__device__ __forceinline__ void g2l_b128(const _Float16* g, _Float16* l) {
    *(half8*)l = *(const half8*)g;
}
__device__ __forceinline__ void wait_async() {}
#endif

// ---------------------------------------------------------------------------
// Kernel 1: x = src + 2D timing signal ; write fp32 master + f16 GEMM copy
// ---------------------------------------------------------------------------
__global__ void k_timing(const float* __restrict__ src,
                         float* __restrict__ x32,
                         _Float16* __restrict__ x16) {
    int idx = blockIdx.x * blockDim.x + threadIdx.x;
    if (idx >= NM * NC) return;
    int c   = idx & 255;
    int s   = (idx >> 8) & 1023;
    int r   = s >> 5;
    int col = s & 31;
    const float LOGK = 0.14620731448716693f;   // ln(10000)/63  (nt=64)
    int   i   = c & 63;
    float inv = __expf(-(float)i * LOGK);
    float pos = (c < 128) ? (float)r : (float)col;
    float ph  = pos * inv;
    float sig = ((c & 64) == 0) ? __sinf(ph) : __cosf(ph);
    float v = src[idx] + sig;
    x32[idx] = v;
    x16[idx] = (_Float16)v;
}

// ---------------------------------------------------------------------------
// Kernel 2: f32 -> f16 conversion (weights)
// ---------------------------------------------------------------------------
__global__ void k_cvt(const float* __restrict__ a, _Float16* __restrict__ o, int n) {
    int i = blockIdx.x * blockDim.x + threadIdx.x;
    if (i < n) o[i] = (_Float16)a[i];
}

// ---------------------------------------------------------------------------
// Helper: assemble a 16x32 f16 A/B WMMA fragment from K-major storage.
// Per ISA layout: lane (m = lane&15, half = lane>>4); elements 0..7 come from
// K = half*8 .. half*8+7, elements 8..15 from K = 16+half*8 .. +7.
// Both runs are 16B contiguous -> two ds/global b128 loads.
// ---------------------------------------------------------------------------
__device__ __forceinline__ v16h mk_frag(const _Float16* p) {
    half8 lo = *(const half8*)(p);
    half8 hi = *(const half8*)(p + 16);
    v16h r;
#pragma unroll
    for (int e = 0; e < 8; ++e) { r[e] = lo[e]; r[e + 8] = hi[e]; }
    return r;
}

// ---------------------------------------------------------------------------
// Kernel 3: generic WMMA GEMM   C[M,N] = A[M,K] * Bw[N,K]^T   (both K-major f16)
// Block: 256 threads = 8 waves; tile 128x128x32; wave computes 64x32.
// Double-buffered LDS, async staged: issue next tile, compute current,
// s_wait_asynccnt + barrier once per k-step.
// outMode: 0 = f32 row-major, 1 = f16 row-major, 2 = f16 "V-transpose"
//          ([b][h][dv][s], s contiguous) for the attention V operand.
// ---------------------------------------------------------------------------
#define BM  128
#define BN  128
#define BK  32
#define LDT 40   // padded LDS row stride (halves): 80B -> spreads 64 banks

__global__ __launch_bounds__(256, 2) void k_gemm(
    const _Float16* __restrict__ A, const _Float16* __restrict__ Bw,
    void* __restrict__ Out, const float* __restrict__ bias,
    int M, int N, int K, float scale, int outMode, int relu)
{
    __shared__ _Float16 Alds[2][BM * LDT];
    __shared__ _Float16 Blds[2][BN * LDT];

    const int bn0  = blockIdx.x * BN;
    const int bm0  = blockIdx.y * BM;
    const int tid  = threadIdx.x;
    const int wave = tid >> 5;
    const int lane = tid & 31;
    const int wm   = (wave & 1) * 64;   // 2 wave strips in M
    const int wn   = (wave >> 1) * 32;  // 4 wave strips in N
    const int lm   = lane & 15;
    const int lh   = lane >> 4;

    const int row = tid >> 1;          // 0..127: tile row this thread stages
    const int seg = (tid & 1) * 16;    // halves offset within the 32-half row

    v8f acc[4][2];
#pragma unroll
    for (int tm = 0; tm < 4; ++tm)
#pragma unroll
        for (int tn = 0; tn < 2; ++tn)
#pragma unroll
            for (int e = 0; e < 8; ++e) acc[tm][tn][e] = 0.0f;

    const _Float16* gArow = A  + (size_t)(bm0 + row) * K + seg;
    const _Float16* gBrow = Bw + (size_t)(bn0 + row) * K + seg;

    // stage tile for k-chunk kc into buffer buf (32B per thread per tile)
    auto stage = [&](int kc, int buf) {
        const _Float16* ga = gArow + kc;
        _Float16* la = Alds[buf] + row * LDT + seg;
        g2l_b128(ga,     la);
        g2l_b128(ga + 8, la + 8);
        const _Float16* gb = gBrow + kc;
        _Float16* lb = Blds[buf] + row * LDT + seg;
        g2l_b128(gb,     lb);
        g2l_b128(gb + 8, lb + 8);
    };

    stage(0, 0);
    wait_async();
    __syncthreads();

    int ib = 0;
    for (int kc = 0; kc < K; kc += BK, ib ^= 1) {
        // prefetch next tile into the other buffer while we compute
        if (kc + BK < K) {
            stage(kc + BK, ib ^ 1);
            __builtin_prefetch(gArow + kc + 2 * BK, 0, 1);
            __builtin_prefetch(gBrow + kc + 2 * BK, 0, 1);
        }

        // ---- fragments + 8 WMMAs per wave for this k-chunk ----
        v16h bfrag[2];
#pragma unroll
        for (int tn = 0; tn < 2; ++tn)
            bfrag[tn] = mk_frag(Blds[ib] + (wn + tn * 16 + lm) * LDT + lh * 8);
#pragma unroll
        for (int tm = 0; tm < 4; ++tm) {
            v16h afrag = mk_frag(Alds[ib] + (wm + tm * 16 + lm) * LDT + lh * 8);
#pragma unroll
            for (int tn = 0; tn < 2; ++tn)
                acc[tm][tn] = __builtin_amdgcn_wmma_f32_16x16x32_f16(
                    false, afrag, false, bfrag[tn], (short)0, acc[tm][tn], false, false);
        }

        wait_async();      // next tile landed in LDS
        __syncthreads();   // all waves done reading current buffer
    }

    // ---- epilogue: D layout = per lane 8 consecutive rows, one column ----
#pragma unroll
    for (int tm = 0; tm < 4; ++tm) {
        int mrun = bm0 + wm + tm * 16 + lh * 8;  // first of 8 consecutive rows
#pragma unroll
        for (int tn = 0; tn < 2; ++tn) {
            int n = bn0 + wn + tn * 16 + lm;
            float bv = bias ? bias[n] : 0.0f;
            float vals[8];
#pragma unroll
            for (int r2 = 0; r2 < 8; ++r2) {
                float v = acc[tm][tn][r2] * scale + bv;
                if (relu) v = fmaxf(v, 0.0f);
                vals[r2] = v;
            }
            if (outMode == 0) {
                float* o = (float*)Out;
#pragma unroll
                for (int r2 = 0; r2 < 8; ++r2)
                    o[(size_t)(mrun + r2) * N + n] = vals[r2];
            } else if (outMode == 1) {
                _Float16* o = (_Float16*)Out;
#pragma unroll
                for (int r2 = 0; r2 < 8; ++r2)
                    o[(size_t)(mrun + r2) * N + n] = (_Float16)vals[r2];
            } else {
                // V-transpose: Out[b][h][dv][s], 8 consecutive s -> one b128 store
                _Float16* o  = (_Float16*)Out;
                int bidx = mrun >> 10, s0 = mrun & 1023;
                int hh = n >> 8, dv = n & 255;
                half8 hv;
#pragma unroll
                for (int r2 = 0; r2 < 8; ++r2) hv[r2] = (_Float16)vals[r2];
                *(half8*)(o + ((size_t)((bidx * NH + hh) * DK + dv)) * NS + s0) = hv;
            }
        }
    }
}

// ---------------------------------------------------------------------------
// Kernel 4: streaming (flash-style) attention.
// 128 threads = 4 waves; each wave owns one 16-row Q block of one (b,h).
// Keys streamed in chunks of 32; probs round-trip LDS for D->A layout.
// ---------------------------------------------------------------------------
#define QSTR 264   // 16B-aligned padded stride (528B) for Q rows in LDS
#define PSTR 40

__global__ __launch_bounds__(128, 1) void k_attn(
    const _Float16* __restrict__ Qh, const _Float16* __restrict__ Kh,
    const _Float16* __restrict__ Vt, _Float16* __restrict__ Oh)
{
    __shared__ _Float16 lds[4 * (16 * QSTR + 16 * PSTR)];
    const int tid  = threadIdx.x;
    const int wave = tid >> 5;
    const int lane = tid & 31;
    _Float16* Qs = lds + wave * (16 * QSTR + 16 * PSTR);
    _Float16* Ps = Qs + 16 * QSTR;

    const int qb   = blockIdx.x * 4 + wave;   // 0 .. 4095
    const int mblk = qb & 63;
    const int bh   = qb >> 6;                 // b*NH + h
    const int b    = bh >> 3;
    const int h    = bh & 7;
    const int m0   = b * NS + mblk * 16;      // global first q row

    // ---- stage cols 0-63 and 128-191 of 16 Q rows (already pre-scaled by 1/sqrt(dk)) into LDS ----
    {
        int qrow = lane >> 1;
        int off  = (lane & 1) * 128;
        const _Float16* g = Qh + (size_t)(m0 + qrow) * HDK + h * DK + off;
        _Float16* l = Qs + qrow * QSTR + off;
#pragma unroll
        for (int i = 0; i < 8; ++i)
            g2l_b128(g + i * 8, l + i * 8);
    }
    wait_async();
    __syncthreads();

    const int lm = lane & 15;
    const int lh = lane >> 4;

    v8f O[16];
#pragma unroll
    for (int dt = 0; dt < 16; ++dt)
#pragma unroll
        for (int e = 0; e < 8; ++e) O[dt][e] = 0.0f;

    float mstate[8], lstate[8];
#pragma unroll
    for (int r = 0; r < 8; ++r) { mstate[r] = -1e30f; lstate[r] = 0.0f; }

    const _Float16* Kbase = Kh + (size_t)b * NS * HDK + h * DK;
    const _Float16* Vbase = Vt + (size_t)bh * DK * NS;

    for (int kc = 0; kc < NS; kc += 32) {
        // ---- scores: two 16x16 tiles, K=256 contraction (16 WMMAs) ----
        v8f sacc[2];
#pragma unroll
        for (int t = 0; t < 2; ++t)
#pragma unroll
            for (int e = 0; e < 8; ++e) sacc[t][e] = 0.0f;

#pragma unroll
        for (int kk = 0; kk < 8; ++kk) {
            v16h qa = mk_frag(Qs + lm * QSTR + kk * 32 + lh * 8);
#pragma unroll
            for (int t = 0; t < 2; ++t) {
                v16h kb = mk_frag(Kbase + (size_t)(kc + t * 16 + lm) * HDK + kk * 32 + lh * 8);
                sacc[t] = __builtin_amdgcn_wmma_f32_16x16x32_f16(
                    false, qa, false, kb, (short)0, sacc[t], false, false);
            }
        }

        // ---- online softmax (rows live in VGPR index, cols across 16 lanes) ----
        float rowmax[8];
#pragma unroll
        for (int r = 0; r < 8; ++r) rowmax[r] = fmaxf(sacc[0][r], sacc[1][r]);
#pragma unroll
        for (int mask = 1; mask < 16; mask <<= 1)
#pragma unroll
            for (int r = 0; r < 8; ++r)
                rowmax[r] = fmaxf(rowmax[r], __shfl_xor(rowmax[r], mask, 32));

        float corr[8], rowsum[8];
#pragma unroll
        for (int r = 0; r < 8; ++r) {
            float mn = fmaxf(mstate[r], rowmax[r]);
            corr[r]  = __expf(mstate[r] - mn);
            mstate[r] = mn;
            rowsum[r] = 0.0f;
        }
#pragma unroll
        for (int t = 0; t < 2; ++t)
#pragma unroll
            for (int r = 0; r < 8; ++r) {
                float p = __expf(sacc[t][r] - mstate[r]);
                rowsum[r] += p;
                Ps[(lh * 8 + r) * PSTR + t * 16 + lm] = (_Float16)p;  // D->A via LDS
            }
#pragma unroll
        for (int mask = 1; mask < 16; mask <<= 1)
#pragma unroll
            for (int r = 0; r < 8; ++r)
                rowsum[r] += __shfl_xor(rowsum[r], mask, 32);
#pragma unroll
        for (int r = 0; r < 8; ++r)
            lstate[r] = lstate[r] * corr[r] + rowsum[r];

        // rescale running output
#pragma unroll
        for (int dt = 0; dt < 16; ++dt)
#pragma unroll
            for (int r = 0; r < 8; ++r) O[dt][r] *= corr[r];

        // ---- P(16x32) @ V(32x256): 16 WMMAs; V B-fragments contiguous in s ----
        v16h pa = mk_frag(Ps + lm * PSTR + lh * 8);
#pragma unroll
        for (int dt = 0; dt < 16; ++dt) {
            v16h vb = mk_frag(Vbase + (size_t)(dt * 16 + lm) * NS + kc + lh * 8);
            O[dt] = __builtin_amdgcn_wmma_f32_16x16x32_f16(
                false, pa, false, vb, (short)0, O[dt], false, false);
        }
    }

    // ---- finalize: divide by row sums, store f16 row-major [M, HDK] ----
    float invl[8];
#pragma unroll
    for (int r = 0; r < 8; ++r) invl[r] = 1.0f / lstate[r];
#pragma unroll
    for (int dt = 0; dt < 16; ++dt) {
        int n = h * DK + dt * 16 + lm;
#pragma unroll
        for (int r = 0; r < 8; ++r) {
            int m = m0 + lh * 8 + r;
            Oh[(size_t)m * HDK + n] = (_Float16)(O[dt][r] * invl[r]);
        }
    }
}

// ---------------------------------------------------------------------------
// Kernel 5: fused residual + layernorm over rows of 256 channels.
// val = alpha*a + bsrc ; out = LN(val)*g + beta. One wave per row.
// ---------------------------------------------------------------------------
__global__ void k_ln(const float* __restrict__ a, const float* __restrict__ bsrc,
                     float alpha, const float* __restrict__ g,
                     const float* __restrict__ beta,
                     float* __restrict__ out32, _Float16* __restrict__ out16)
{
    int row  = blockIdx.x * 8 + (threadIdx.x >> 5);
    int lane = threadIdx.x & 31;
    float v[8], sum = 0.0f, sq = 0.0f;
#pragma unroll
    for (int i = 0; i < 8; ++i) {
        int c = lane + i * 32;
        float t = alpha * a[(size_t)row * NC + c] + bsrc[(size_t)row * NC + c];
        v[i] = t; sum += t; sq += t * t;
    }
#pragma unroll
    for (int mask = 1; mask < 32; mask <<= 1) {
        sum += __shfl_xor(sum, mask, 32);
        sq  += __shfl_xor(sq,  mask, 32);
    }
    float mean = sum * (1.0f / 256.0f);
    float var  = sq * (1.0f / 256.0f) - mean * mean;
    float rs   = rsqrtf(var + 1e-5f);
#pragma unroll
    for (int i = 0; i < 8; ++i) {
        int c = lane + i * 32;
        float y = (v[i] - mean) * rs * g[c] + beta[c];
        out32[(size_t)row * NC + c] = y;
        if (out16) out16[(size_t)row * NC + c] = (_Float16)y;
    }
}

// ---------------------------------------------------------------------------
// Host launcher
// ---------------------------------------------------------------------------
extern "C" void kernel_launch(void* const* d_in, const int* in_sizes, int n_in,
                              void* d_out, int out_size, void* d_ws, size_t ws_size,
                              hipStream_t stream) {
    const float* src    = (const float*)d_in[0];
    const float* w_q    = (const float*)d_in[1];
    const float* w_k    = (const float*)d_in[2];
    const float* w_v    = (const float*)d_in[3];
    const float* w_o    = (const float*)d_in[4];
    const float* ffn1_w = (const float*)d_in[5];
    const float* ffn1_b = (const float*)d_in[6];
    const float* ffn2_w = (const float*)d_in[7];
    const float* ffn2_b = (const float*)d_in[8];
    const float* ln1_g  = (const float*)d_in[9];
    const float* ln1_b  = (const float*)d_in[10];
    const float* ln2_g  = (const float*)d_in[11];
    const float* ln2_b  = (const float*)d_in[12];

    char* wsp = (char*)d_ws;
    auto alloc = [&](size_t bytes) -> void* {
        void* p = (void*)wsp;
        wsp += (bytes + 255) & ~(size_t)255;
        return p;
    };
    float*     x32  = (float*)    alloc((size_t)NM * NC * 4);
    _Float16*  x16  = (_Float16*) alloc((size_t)NM * NC * 2);
    _Float16*  wq16 = (_Float16*) alloc((size_t)HDK * NC * 2);
    _Float16*  wk16 = (_Float16*) alloc((size_t)HDK * NC * 2);
    _Float16*  wv16 = (_Float16*) alloc((size_t)HDK * NC * 2);
    _Float16*  wo16 = (_Float16*) alloc((size_t)NC * HDK * 2);
    _Float16*  w116 = (_Float16*) alloc((size_t)DFF * NC * 2);
    _Float16*  w216 = (_Float16*) alloc((size_t)NC * DFF * 2);
    _Float16*  q16  = (_Float16*) alloc((size_t)NM * HDK * 2);
    _Float16*  k16  = (_Float16*) alloc((size_t)NM * HDK * 2);
    _Float16*  vt16 = (_Float16*) alloc((size_t)NM * HDK * 2);  // [b][h][dv][s]
    _Float16*  o16  = (_Float16*) alloc((size_t)NM * HDK * 2);
    float*     P32  = (float*)    alloc((size_t)NM * NC * 4);
    float*     h32  = (float*)    alloc((size_t)NM * NC * 4);
    _Float16*  h16  = (_Float16*) alloc((size_t)NM * NC * 2);
    _Float16*  g16  = (_Float16*) alloc((size_t)NM * DFF * 2);
    float*     F32  = (float*)    alloc((size_t)NM * NC * 4);

    // 1. x = src + timing signal
    k_timing<<<(NM * NC) / 256, 256, 0, stream>>>(src, x32, x16);

    // 2. weights -> f16
    const int WN = HDK * NC;  // 524288 for all six weights
    k_cvt<<<WN / 256, 256, 0, stream>>>(w_q,    wq16, WN);
    k_cvt<<<WN / 256, 256, 0, stream>>>(w_k,    wk16, WN);
    k_cvt<<<WN / 256, 256, 0, stream>>>(w_v,    wv16, WN);
    k_cvt<<<WN / 256, 256, 0, stream>>>(w_o,    wo16, WN);
    k_cvt<<<WN / 256, 256, 0, stream>>>(ffn1_w, w116, WN);
    k_cvt<<<WN / 256, 256, 0, stream>>>(ffn2_w, w216, WN);

    // 3. QKV projections (Q pre-scaled by 1/sqrt(256) = 1/16)
    dim3 gQKV(HDK / BN, NM / BM);
    k_gemm<<<gQKV, 256, 0, stream>>>(x16, wq16, (void*)q16,  nullptr, NM, HDK, NC, 0.0625f, 1, 0);
    k_gemm<<<gQKV, 256, 0, stream>>>(x16, wk16, (void*)k16,  nullptr, NM, HDK, NC, 1.0f,    1, 0);
    k_gemm<<<gQKV, 256, 0, stream>>>(x16, wv16, (void*)vt16, nullptr, NM, HDK, NC, 1.0f,    2, 0);

    // 4. streaming attention -> o16 [M, HDK]
    k_attn<<<(NB * NH * (NS / 16)) / 4, 128, 0, stream>>>(q16, k16, vt16, o16);

    // 5. output projection P = o @ w_o^T  (f32)
    dim3 gP(NC / BN, NM / BM);
    k_gemm<<<gP, 256, 0, stream>>>(o16, wo16, (void*)P32, nullptr, NM, NC, HDK, 1.0f, 0, 0);

    // 6. h = LN(2*x + P)   (torch code adds x twice: x + (proj + xs))
    k_ln<<<NM / 8, 256, 0, stream>>>(x32, P32, 2.0f, ln1_g, ln1_b, h32, h16);

    // 7. FFN1: relu(h @ W1^T + b1) -> f16
    dim3 gF1(DFF / BN, NM / BM);
    k_gemm<<<gF1, 256, 0, stream>>>(h16, w116, (void*)g16, ffn1_b, NM, DFF, NC, 1.0f, 1, 1);

    // 8. FFN2: g @ W2^T + b2 -> f32
    dim3 gF2(NC / BN, NM / BM);
    k_gemm<<<gF2, 256, 0, stream>>>(g16, w216, (void*)F32, ffn2_b, NM, NC, DFF, 1.0f, 0, 0);

    // 9. out = LN(h + F)
    k_ln<<<NM / 8, 256, 0, stream>>>(h32, F32, 1.0f, ln2_g, ln2_b, (float*)d_out, nullptr);
}
